// PTV3_CPE_214748364939
// MI455X (gfx1250) — matrix operand order repaired and checked by reading, not compiled, not run on hardware
//
#include <hip/hip_runtime.h>
#include <hip/hip_bf16.h>

typedef __attribute__((ext_vector_type(16))) _Float16 v16h;
typedef __attribute__((ext_vector_type(8)))  _Float16 v8h;
typedef __attribute__((ext_vector_type(8)))  float    v8f;

union V16 { v16h v; v8h h[2]; };

#define C_CH 128
#define K_TAPS 27
#define EPSV 1e-5f

// ---- CDNA5 async global->LDS copy (ASYNCcnt-tracked), 16B per lane ----
__device__ __forceinline__ void async_ld_b128(const void* g, void* l) {
    unsigned lofs = (unsigned)(unsigned long long)(__attribute__((address_space(3))) void*)l;
    unsigned long long ga = (unsigned long long)g;
    asm volatile("global_load_async_to_lds_b128 %0, %1, off"
                 :: "v"(lofs), "v"(ga) : "memory");
}

__device__ __forceinline__ void wait_async0() {
#if __has_builtin(__builtin_amdgcn_s_wait_asynccnt)
    __builtin_amdgcn_s_wait_asynccnt(0);
#else
    asm volatile("s_wait_asynccnt 0" ::: "memory");
#endif
}

// ---------------- prep kernels ----------------

__global__ void cvt_feats_f16(const float* __restrict__ f, _Float16* __restrict__ o, int n) {
    int i = blockIdx.x * blockDim.x + threadIdx.x;
    if (i < n) o[i] = (_Float16)f[i];
}

// Pack conv_w [27][128][128] (cin rows, cout cols) into WMMA B-fragment order:
// frag index = (((k*4 + ck)*8 + nt)*32 + lane)*16 + h
// lane<16: col=lane, K rows 0..15 of chunk; lane>=16: col=lane-16, K rows 16..31.
__global__ void pack_conv_w(const float* __restrict__ w, _Float16* __restrict__ dst, int total) {
    int t = blockIdx.x * blockDim.x + threadIdx.x;
    if (t >= total) return;
    int co = t & 127;
    int ci = (t >> 7) & 127;
    int k  = t >> 14;
    int ck = ci >> 5, kk = ci & 31;
    int nt = co >> 4, cl = co & 15;
    int lane = cl + (kk & 16);
    int h = kk & 15;
    dst[(((k * 4 + ck) * 8 + nt) * 32 + lane) * 16 + h] = (_Float16)w[t];
}

// lin_w is [out,in]; y = x @ W^T  =>  B[i][o] = lin_w[o][i]; same frag layout (K dim = i).
__global__ void pack_lin_w(const float* __restrict__ w, _Float16* __restrict__ dst, int total) {
    int t = blockIdx.x * blockDim.x + threadIdx.x;
    if (t >= total) return;
    int i  = t & 127;          // in  (K dim)
    int o  = t >> 7;           // out (N dim)
    int ck = i >> 5, kk = i & 31;
    int nt = o >> 4, cl = o & 15;
    int lane = cl + (kk & 16);
    int h = kk & 15;
    dst[((ck * 8 + nt) * 32 + lane) * 16 + h] = (_Float16)w[t];
}

// ---------------- fused main kernel ----------------
// 8 waves * 32 rows = 256 rows per block.
// smem halves: [0,16384) convB buf0 | [16384,32768) convB buf1 | [32768,49152) lin B
// After the conv loop, [0,32768) is reused as 8 waves * 4096-half transpose scratch.

__global__ void __launch_bounds__(256)
ptv3_fused(const int*   __restrict__ nbi,
           const float* __restrict__ conv_b,
           const float* __restrict__ lin_b,
           const float* __restrict__ ln_w,
           const float* __restrict__ ln_b,
           const _Float16* __restrict__ featsh,
           const _Float16* __restrict__ Bpre,
           const _Float16* __restrict__ Lpre,
           float* __restrict__ out, int N) {
    extern __shared__ __align__(16) _Float16 smem[];
    const int tid  = threadIdx.x;
    const int lane = tid & 31;
    const int wave = tid >> 5;
    const int m    = lane & 15;      // row-in-tile (A) / col (B,C,D)
    const int hig  = lane >> 4;      // half-wave select
    const int rowBase = (blockIdx.x * 8 + wave) * 32;

    // Async-stage Linear B fragments (32 KB) and conv tap 0 (32 KB): 128 B per thread each.
    {
        const char* ls = (const char*)Lpre + tid * 128;
        const char* bs = (const char*)Bpre + tid * 128;
        char* ld = (char*)(smem + 32768) + tid * 128;
        char* bd = (char*)smem + tid * 128;
#pragma unroll
        for (int j = 0; j < 8; ++j) {
            async_ld_b128(ls + j * 16, ld + j * 16);
            async_ld_b128(bs + j * 16, bd + j * 16);
        }
    }
    wait_async0();
    __syncthreads();

    // Gather index rows for this wave's two 16-row tiles (clamped; stores masked later).
    int rg0 = rowBase + m;      if (rg0 > N - 1) rg0 = N - 1;
    int rg1 = rowBase + 16 + m; if (rg1 > N - 1) rg1 = N - 1;
    const int* idxrow0 = nbi + rg0 * K_TAPS;
    const int* idxrow1 = nbi + rg1 * K_TAPS;

    v8f acc[2][8] = {};

    for (int k = 0; k < K_TAPS; ++k) {
        // Async DMA of tap k+1 weights into the other LDS buffer (last read at k-1, barrier-safe).
        if (k + 1 < K_TAPS) {
            const char* src = (const char*)(Bpre + (k + 1) * (C_CH * C_CH)) + tid * 128;
            char* dst = (char*)(smem + ((k + 1) & 1) * 16384) + tid * 128;
#pragma unroll
            for (int j = 0; j < 8; ++j) async_ld_b128(src + j * 16, dst + j * 16);
        }
        const _Float16* buf = smem + (k & 1) * 16384;
        const _Float16* ar0 = featsh + (size_t)idxrow0[k] * C_CH;
        const _Float16* ar1 = featsh + (size_t)idxrow1[k] * C_CH;
#pragma unroll
        for (int ck = 0; ck < 4; ++ck) {
            const int kb = ck * 32 + hig * 16;
            V16 A0, A1;
            A0.h[0] = *(const v8h*)(ar0 + kb);
            A0.h[1] = *(const v8h*)(ar0 + kb + 8);
            A1.h[0] = *(const v8h*)(ar1 + kb);
            A1.h[1] = *(const v8h*)(ar1 + kb + 8);
#pragma unroll
            for (int nt = 0; nt < 8; ++nt) {
                const v16h B = *(const v16h*)(buf + ((ck * 8 + nt) * 32 + lane) * 16);
                acc[0][nt] = __builtin_amdgcn_wmma_f32_16x16x32_f16(
                    false, A0.v, false, B, (short)0, acc[0][nt], false, false);
                acc[1][nt] = __builtin_amdgcn_wmma_f32_16x16x32_f16(
                    false, A1.v, false, B, (short)0, acc[1][nt], false, false);
            }
        }
        wait_async0();       // my async writes for tap k+1 landed in LDS
        __syncthreads();     // everyone's writes visible; buffer reads done block-wide
    }

    // conv bias, then transpose C/D layout -> A layout in the (now free) conv buffers.
    const int rofs = hig * 8;
    _Float16* xp = smem + wave * 4096;   // 32 rows * 128 halves
#pragma unroll
    for (int nt = 0; nt < 8; ++nt) {
        const float cb = conv_b[nt * 16 + m];
#pragma unroll
        for (int r = 0; r < 8; ++r) {
            xp[(rofs + r) * C_CH + nt * 16 + m]        = (_Float16)(acc[0][nt][r] + cb);
            xp[(16 + rofs + r) * C_CH + nt * 16 + m]   = (_Float16)(acc[1][nt][r] + cb);
        }
    }
    __syncthreads();

    // Linear: lacc = conv_out @ lin_w^T
    v8f lacc[2][8] = {};
    const _Float16* lbuf = smem + 32768;
    const _Float16* xr0  = xp + m * C_CH;
    const _Float16* xr1  = xp + (16 + m) * C_CH;
#pragma unroll
    for (int ck = 0; ck < 4; ++ck) {
        const int kb = ck * 32 + hig * 16;
        V16 A0, A1;
        A0.h[0] = *(const v8h*)(xr0 + kb);
        A0.h[1] = *(const v8h*)(xr0 + kb + 8);
        A1.h[0] = *(const v8h*)(xr1 + kb);
        A1.h[1] = *(const v8h*)(xr1 + kb + 8);
#pragma unroll
        for (int nt = 0; nt < 8; ++nt) {
            const v16h B = *(const v16h*)(lbuf + ((ck * 8 + nt) * 32 + lane) * 16);
            lacc[0][nt] = __builtin_amdgcn_wmma_f32_16x16x32_f16(
                false, A0.v, false, B, (short)0, lacc[0][nt], false, false);
            lacc[1][nt] = __builtin_amdgcn_wmma_f32_16x16x32_f16(
                false, A1.v, false, B, (short)0, lacc[1][nt], false, false);
        }
    }

    // LayerNorm over the 128 channels of each row + store.
    float lb[8], lwv[8], lbv[8];
#pragma unroll
    for (int nt = 0; nt < 8; ++nt) {
        lb[nt]  = lin_b[nt * 16 + m];
        lwv[nt] = ln_w[nt * 16 + m];
        lbv[nt] = ln_b[nt * 16 + m];
    }
#pragma unroll
    for (int mf = 0; mf < 2; ++mf) {
#pragma unroll
        for (int r = 0; r < 8; ++r) {
            float s = 0.f, q = 0.f;
#pragma unroll
            for (int nt = 0; nt < 8; ++nt) {
                const float v = lacc[mf][nt][r] + lb[nt];
                s += v; q += v * v;
            }
            // butterfly over the 16 lanes holding this row (halves stay independent)
#pragma unroll
            for (int off = 1; off < 16; off <<= 1) {
                s += __shfl_xor(s, off, 32);
                q += __shfl_xor(q, off, 32);
            }
            const float mu  = s * (1.0f / 128.0f);
            const float var = q * (1.0f / 128.0f) - mu * mu;
            const float inv = rsqrtf(var + EPSV);
            const int row = rowBase + mf * 16 + rofs + r;
            if (row < N) {
                float* orow = out + (size_t)row * C_CH;
#pragma unroll
                for (int nt = 0; nt < 8; ++nt) {
                    const float v = lacc[mf][nt][r] + lb[nt];
                    orow[nt * 16 + m] = (v - mu) * inv * lwv[nt] + lbv[nt];
                }
            }
        }
    }
}

// ---------------- launch ----------------

extern "C" void kernel_launch(void* const* d_in, const int* in_sizes, int n_in,
                              void* d_out, int out_size, void* d_ws, size_t ws_size,
                              hipStream_t stream) {
    (void)n_in; (void)out_size; (void)ws_size;
    const float* feats  = (const float*)d_in[0];
    const int*   nbi    = (const int*)d_in[1];
    const float* conv_w = (const float*)d_in[2];
    const float* conv_b = (const float*)d_in[3];
    const float* lin_w  = (const float*)d_in[4];
    const float* lin_b  = (const float*)d_in[5];
    const float* ln_w   = (const float*)d_in[6];
    const float* ln_b   = (const float*)d_in[7];
    float* out = (float*)d_out;

    const int N  = in_sizes[0] / C_CH;
    const int ne = N * C_CH;

    _Float16* featsh = (_Float16*)d_ws;
    _Float16* Bpre   = featsh + (size_t)ne;                 // 27*128*128 halves
    _Float16* Lpre   = Bpre + (size_t)K_TAPS * C_CH * C_CH; // 128*128 halves

    cvt_feats_f16<<<(ne + 255) / 256, 256, 0, stream>>>(feats, featsh, ne);
    pack_conv_w<<<(K_TAPS * C_CH * C_CH + 255) / 256, 256, 0, stream>>>(
        conv_w, Bpre, K_TAPS * C_CH * C_CH);
    pack_lin_w<<<(C_CH * C_CH + 255) / 256, 256, 0, stream>>>(lin_w, Lpre, C_CH * C_CH);

    const int blocks = (N + 255) / 256;                 // 256 rows per block
    const size_t shmem = 49152 * sizeof(_Float16);      // 96 KB -> 3 blocks per WGP
    ptv3_fused<<<blocks, 256, shmem, stream>>>(
        nbi, conv_b, lin_b, ln_w, ln_b, featsh, Bpre, Lpre, out, N);
}
